// GraphDiT_88115549045157
// MI455X (gfx1250) — matrix operand order repaired and checked by reading, not compiled, not run on hardware
//
#include <hip/hip_runtime.h>
#include <hip/hip_bf16.h>

typedef _Float16 v16h __attribute__((ext_vector_type(16)));
typedef _Float16 v8h  __attribute__((ext_vector_type(8)));
typedef float    v8f  __attribute__((ext_vector_type(8)));

#define BB     2
#define NATOM  48
#define ATYPES 16
#define ETYPES 5
#define HIDN   384
#define NHEAD  8
#define HD     48
#define NLYR   4
#define MLPN   1536
#define LSEQ   (NATOM + NATOM * NATOM)   /* 2352 */
#define ROWS   (BB * LSEQ)               /* 4704 */

union AFrag { v16h v; v8h h[2]; };

static __device__ inline v8f wmma_f16(v16h a, v16h b, v8f c) {
  return __builtin_amdgcn_wmma_f32_16x16x32_f16(false, a, false, b, (short)0, c,
                                                false, false);
}

static __device__ inline float redmax16(float v) {
  v = fmaxf(v, __shfl_xor(v, 1, 32));
  v = fmaxf(v, __shfl_xor(v, 2, 32));
  v = fmaxf(v, __shfl_xor(v, 4, 32));
  v = fmaxf(v, __shfl_xor(v, 8, 32));
  return v;
}
static __device__ inline float redsum16(float v) {
  v += __shfl_xor(v, 1, 32);
  v += __shfl_xor(v, 2, 32);
  v += __shfl_xor(v, 4, 32);
  v += __shfl_xor(v, 8, 32);
  return v;
}

// ---------------------------------------------------------------------------
// Pack f32 weight [K, N] row-major into WMMA B-fragment layout (f16).
// dst[((kc*ntiles+nt)*32 + lane)*16 + p] = W[kc*32 + (lane>=16?16:0) + p, nt*16 + (lane&15)]
// ---------------------------------------------------------------------------
__global__ __launch_bounds__(32) void k_pack(const float* __restrict__ W,
                                             _Float16* __restrict__ out, int Ncols) {
  int kc = blockIdx.x, nt = blockIdx.y, lane = threadIdx.x;
  int ntiles = Ncols >> 4;
  int col = (nt << 4) + (lane & 15);
  int kb = (kc << 5) + ((lane >= 16) ? 16 : 0);
  _Float16* dst = out + ((size_t)(kc * ntiles + nt) * 32 + lane) * 16;
  for (int p = 0; p < 16; ++p)
    dst[p] = (_Float16)W[(size_t)(kb + p) * Ncols + col];
}

// ---------------------------------------------------------------------------
// WMMA GEMM, register-blocked 32x64 per wave (2 m-tiles x 4 n-tiles):
//   C = A(f16, MxK row-major) @ Bpacked + bias
// mode 0: outH = f16(C)          (q/k/v projections)
// mode 1: outF = res + C         (o-proj / mlp2 residual add, res may alias outF)
// mode 2: outH = f16(gelu(C))    (mlp1, exact GELU)
// grid = (N/64, M/32).  Each K-chunk: 2 A-frags + 4 B-frags feed 8 WMMAs.
// ---------------------------------------------------------------------------
__global__ __launch_bounds__(32) void k_gemm(const _Float16* __restrict__ A,
                                             const _Float16* __restrict__ Bp,
                                             const float* __restrict__ bias,
                                             const float* __restrict__ res,
                                             float* __restrict__ outF,
                                             _Float16* __restrict__ outH,
                                             int K, int Ncols, int mode) {
  int lane = threadIdx.x;
  int l15 = lane & 15;
  int nt0 = blockIdx.x << 2;          // first of 4 n-tiles
  int mt0 = blockIdx.y << 1;          // first of 2 m-tiles
  int ntiles = Ncols >> 4;
  int klo = (lane >= 16) ? 16 : 0;
  const _Float16* arow0 = A + (size_t)((mt0 << 4) + l15) * K;
  const _Float16* arow1 = arow0 + (size_t)16 * K;

  v8f acc[2][4];
  for (int i = 0; i < 2; ++i)
    for (int j = 0; j < 4; ++j)
      for (int e = 0; e < 8; ++e) acc[i][j][e] = 0.0f;

  int kchunks = K >> 5;
  for (int kc = 0; kc < kchunks; ++kc) {
    int kb = kc << 5;
    AFrag a0, a1;
    a0.h[0] = *(const v8h*)(arow0 + kb + klo);
    a0.h[1] = *(const v8h*)(arow0 + kb + klo + 8);
    a1.h[0] = *(const v8h*)(arow1 + kb + klo);
    a1.h[1] = *(const v8h*)(arow1 + kb + klo + 8);
    const _Float16* brow = Bp + ((size_t)(kc * ntiles + nt0) * 32 + lane) * 16;
#pragma unroll
    for (int j = 0; j < 4; ++j) {
      v16h b = *(const v16h*)(brow + (size_t)j * 32 * 16);
      acc[0][j] = wmma_f16(a0.v, b, acc[0][j]);
      acc[1][j] = wmma_f16(a1.v, b, acc[1][j]);
    }
  }

  int rsel = (lane >= 16) ? 8 : 0;
#pragma unroll
  for (int i = 0; i < 2; ++i) {
    int rbase = ((mt0 + i) << 4) + rsel;
#pragma unroll
    for (int j = 0; j < 4; ++j) {
      int col = ((nt0 + j) << 4) + l15;
      float bv = bias[col];
#pragma unroll
      for (int v = 0; v < 8; ++v) {
        size_t idx = (size_t)(rbase + v) * Ncols + col;
        float val = acc[i][j][v] + bv;
        if (mode == 0) {
          outH[idx] = (_Float16)val;
        } else if (mode == 1) {
          outF[idx] = res[idx] + val;
        } else {
          outH[idx] = (_Float16)(0.5f * val * (1.0f + erff(val * 0.70710678118654752f)));
        }
      }
    }
  }
}

// ---------------------------------------------------------------------------
// Flash attention: one wave per (qtile, head, batch).  32 keys per iteration.
// Q/K/V are f16 [B, LSEQ, HIDN]; head h occupies cols [h*48, h*48+48).
// ---------------------------------------------------------------------------
__global__ __launch_bounds__(32) void k_attn(const _Float16* __restrict__ Qm,
                                             const _Float16* __restrict__ Km,
                                             const _Float16* __restrict__ Vm,
                                             _Float16* __restrict__ Om) {
  __shared__ __align__(32) _Float16 ldsK[32 * 48];   // [key][d]
  __shared__ __align__(32) _Float16 ldsVt[48 * 32];  // [d][key]
  __shared__ __align__(32) _Float16 ldsP[16 * 32];   // [qrow][key]

  int lane = threadIdx.x;
  int qt = blockIdx.x, hh = blockIdx.y, b = blockIdx.z;
  size_t base = (size_t)b * LSEQ * HIDN + hh * HD;
  int klo = (lane >= 16) ? 16 : 0;
  int l15 = lane & 15;

  // Q fragments: chunk0 covers d 0..31, chunk1 covers d 32..47 (+16 zero pad)
  int qrow = qt * 16 + l15;
  const _Float16* qp = Qm + base + (size_t)qrow * HIDN;
  AFrag qa0, qa1;
  qa0.h[0] = *(const v8h*)(qp + klo);
  qa0.h[1] = *(const v8h*)(qp + klo + 8);
  if (lane < 16) { qa1.h[0] = *(const v8h*)(qp + 32); qa1.h[1] = *(const v8h*)(qp + 40); }
  else { v8h z; for (int i = 0; i < 8; ++i) z[i] = (_Float16)0; qa1.h[0] = z; qa1.h[1] = z; }

  float m_i[8], l_i[8];
  v8f o0, o1, o2;
  for (int v = 0; v < 8; ++v) {
    m_i[v] = -1e30f; l_i[v] = 0.0f; o0[v] = 0.0f; o1[v] = 0.0f; o2[v] = 0.0f;
  }
  const float inv_scale = 0.14433756729740643f;  // 1/sqrt(48)

  int nkb = (LSEQ + 31) / 32;  // 74
  for (int kb = 0; kb < nkb; ++kb) {
    int kbase = kb * 32;
    int key = kbase + lane;
    if (key < LSEQ) {
      const _Float16* kp = Km + base + (size_t)key * HIDN;
      const _Float16* vp = Vm + base + (size_t)key * HIDN;
      v16h kA = *(const v16h*)(kp);
      v16h kB = *(const v16h*)(kp + 16);
      v16h kC = *(const v16h*)(kp + 32);
      *(v16h*)(&ldsK[lane * 48])      = kA;
      *(v16h*)(&ldsK[lane * 48 + 16]) = kB;
      *(v16h*)(&ldsK[lane * 48 + 32]) = kC;
      v16h vA = *(const v16h*)(vp);
      v16h vB = *(const v16h*)(vp + 16);
      v16h vC = *(const v16h*)(vp + 32);
      for (int d = 0; d < 16; ++d) {
        ldsVt[d * 32 + lane]        = vA[d];
        ldsVt[(d + 16) * 32 + lane] = vB[d];
        ldsVt[(d + 32) * 32 + lane] = vC[d];
      }
    } else {
      for (int d = 0; d < 48; ++d) {
        ldsK[lane * 48 + d] = (_Float16)0;
        ldsVt[d * 32 + lane] = (_Float16)0;
      }
    }
    __syncthreads();

    // K B-fragments (two 16-key tiles x two d-chunks)
    int coff = (lane >= 16) ? 16 : 0;
    v16h kb0c0 = *(const v16h*)(&ldsK[l15 * 48 + coff]);
    v16h kb1c0 = *(const v16h*)(&ldsK[(16 + l15) * 48 + coff]);
    v16h kb0c1, kb1c1;
    if (lane < 16) {
      kb0c1 = *(const v16h*)(&ldsK[l15 * 48 + 32]);
      kb1c1 = *(const v16h*)(&ldsK[(16 + l15) * 48 + 32]);
    } else {
      for (int i = 0; i < 16; ++i) { kb0c1[i] = (_Float16)0; kb1c1[i] = (_Float16)0; }
    }

    v8f s0, s1;
    for (int i = 0; i < 8; ++i) { s0[i] = 0.0f; s1[i] = 0.0f; }
    s0 = wmma_f16(qa0.v, kb0c0, s0);
    s0 = wmma_f16(qa1.v, kb0c1, s0);
    s1 = wmma_f16(qa0.v, kb1c0, s1);
    s1 = wmma_f16(qa1.v, kb1c1, s1);

    bool ok0 = (kbase + l15) < LSEQ;
    bool ok1 = (kbase + 16 + l15) < LSEQ;

    float p0a[8], p1a[8], alpha[8];
    for (int v = 0; v < 8; ++v) {
      float a0 = ok0 ? s0[v] * inv_scale : -1e30f;
      float a1 = ok1 ? s1[v] * inv_scale : -1e30f;
      float mt = redmax16(fmaxf(a0, a1));
      float mnew = fmaxf(m_i[v], mt);
      float al = expf(m_i[v] - mnew);
      float p0 = expf(a0 - mnew);
      float p1 = expf(a1 - mnew);
      float rs = redsum16(p0 + p1);
      l_i[v] = l_i[v] * al + rs;
      m_i[v] = mnew;
      alpha[v] = al;
      p0a[v] = p0; p1a[v] = p1;
    }
    for (int v = 0; v < 8; ++v) { o0[v] *= alpha[v]; o1[v] *= alpha[v]; o2[v] *= alpha[v]; }

    // stage P (16x32, f16) to LDS, reload as A-fragment
    int rsel = (lane >= 16) ? 8 : 0;
    for (int v = 0; v < 8; ++v) {
      int pr = v + rsel;
      ldsP[pr * 32 + l15]      = (_Float16)p0a[v];
      ldsP[pr * 32 + 16 + l15] = (_Float16)p1a[v];
    }
    __syncthreads();

    AFrag pf;
    pf.h[0] = *(const v8h*)(&ldsP[l15 * 32 + klo]);
    pf.h[1] = *(const v8h*)(&ldsP[l15 * 32 + klo + 8]);

    // V B-fragments per 16-wide d tile (contiguous in transposed LDS)
    v16h vb0 = *(const v16h*)(&ldsVt[l15 * 32 + coff]);
    v16h vb1 = *(const v16h*)(&ldsVt[(16 + l15) * 32 + coff]);
    v16h vb2 = *(const v16h*)(&ldsVt[(32 + l15) * 32 + coff]);
    o0 = wmma_f16(pf.v, vb0, o0);
    o1 = wmma_f16(pf.v, vb1, o1);
    o2 = wmma_f16(pf.v, vb2, o2);
    __syncthreads();
  }

  int rsel = (lane >= 16) ? 8 : 0;
  for (int v = 0; v < 8; ++v) {
    float inv_l = 1.0f / l_i[v];
    size_t rb = base + (size_t)(qt * 16 + v + rsel) * HIDN;
    Om[rb + l15]      = (_Float16)(o0[v] * inv_l);
    Om[rb + 16 + l15] = (_Float16)(o1[v] * inv_l);
    Om[rb + 32 + l15] = (_Float16)(o2[v] * inv_l);
  }
}

// ---------------------------------------------------------------------------
// Embedding: h[b,l,:] = (node or edge proj) + pos
// ---------------------------------------------------------------------------
__global__ void k_embed(const float* __restrict__ x, const float* __restrict__ adj,
                        const float* __restrict__ node_w, const float* __restrict__ node_b,
                        const float* __restrict__ edge_w, const float* __restrict__ edge_b,
                        const float* __restrict__ pos, float* __restrict__ h) {
  int idx = blockIdx.x * blockDim.x + threadIdx.x;
  if (idx >= ROWS * HIDN) return;
  int col = idx % HIDN;
  int row = idx / HIDN;
  int b = row / LSEQ, l = row % LSEQ;
  float acc;
  if (l < NATOM) {
    const float* xp = x + (size_t)(b * NATOM + l) * ATYPES;
    acc = node_b[col];
    for (int a = 0; a < ATYPES; ++a) acc += xp[a] * node_w[a * HIDN + col];
  } else {
    const float* ap = adj + ((size_t)b * NATOM * NATOM + (l - NATOM)) * ETYPES;
    acc = edge_b[col];
    for (int e = 0; e < ETYPES; ++e) acc += ap[e] * edge_w[e * HIDN + col];
  }
  h[idx] = acc + pos[(size_t)l * HIDN + col];
}

// Sinusoidal timestep embedding
__global__ void k_temb(const float* __restrict__ t, float* __restrict__ temb) {
  int idx = blockIdx.x * blockDim.x + threadIdx.x;
  if (idx >= BB * HIDN) return;
  int b = idx / HIDN, i = idx % HIDN;
  const int half = HIDN / 2;
  if (i < half) {
    float f = expf(-9.210340371976184f * (float)i / (float)half);
    temb[idx] = sinf(t[b] * f);
  } else {
    float f = expf(-9.210340371976184f * (float)(i - half) / (float)half);
    temb[idx] = cosf(t[b] * f);
  }
}

__global__ void k_tmlp1(const float* __restrict__ temb, const float* __restrict__ w,
                        const float* __restrict__ bias, float* __restrict__ out) {
  int idx = blockIdx.x * blockDim.x + threadIdx.x;
  if (idx >= BB * MLPN) return;
  int b = idx / MLPN, j = idx % MLPN;
  float acc = bias[j];
  const float* tp = temb + b * HIDN;
  for (int i = 0; i < HIDN; ++i) acc += tp[i] * w[(size_t)i * MLPN + j];
  out[idx] = acc / (1.0f + expf(-acc));  // SiLU
}

__global__ void k_tmlp2(const float* __restrict__ hmid, const float* __restrict__ w,
                        const float* __restrict__ bias, float* __restrict__ out) {
  int idx = blockIdx.x * blockDim.x + threadIdx.x;
  if (idx >= BB * HIDN) return;
  int b = idx / HIDN, j = idx % HIDN;
  float acc = bias[j];
  const float* hp = hmid + b * MLPN;
  for (int i = 0; i < MLPN; ++i) acc += hp[i] * w[(size_t)i * HIDN + j];
  out[idx] = acc;
}

// AdaLN modulation for all layers/both norms: scsh[layer][which][b][768]
__global__ void k_ada(const float* __restrict__ cond,
                      const float* __restrict__ w1, const float* __restrict__ b1,
                      const float* __restrict__ w2, const float* __restrict__ b2,
                      float* __restrict__ scsh) {
  int idx = blockIdx.x * blockDim.x + threadIdx.x;
  if (idx >= NLYR * 2 * BB * 2 * HIDN) return;
  int j = idx % (2 * HIDN);
  int b = (idx / (2 * HIDN)) % BB;
  int which = (idx / (2 * HIDN * BB)) % 2;
  int layer = idx / (2 * HIDN * BB * 2);
  const float* w = (which == 0) ? w1 : w2;
  const float* bs = (which == 0) ? b1 : b2;
  float acc = bs[layer * 2 * HIDN + j];
  const float* cp = cond + b * HIDN;
  for (int k = 0; k < HIDN; ++k)
    acc += cp[k] * w[((size_t)layer * HIDN + k) * (2 * HIDN) + j];
  scsh[idx] = acc;
}

// LayerNorm (no affine) * (1+sc) + sh -> f16
__global__ __launch_bounds__(128) void k_lnmod(const float* __restrict__ h,
                                               const float* __restrict__ scsh,
                                               _Float16* __restrict__ out) {
  __shared__ float rs[128], rq[128];
  int row = blockIdx.x, tid = threadIdx.x;
  const float* x = h + (size_t)row * HIDN;
  float s = 0.0f, s2 = 0.0f;
  for (int c = tid; c < HIDN; c += 128) { float v = x[c]; s += v; s2 += v * v; }
  rs[tid] = s; rq[tid] = s2;
  __syncthreads();
  for (int o = 64; o > 0; o >>= 1) {
    if (tid < o) { rs[tid] += rs[tid + o]; rq[tid] += rq[tid + o]; }
    __syncthreads();
  }
  float mean = rs[0] / (float)HIDN;
  float var = rq[0] / (float)HIDN - mean * mean;
  float rstd = rsqrtf(var + 1e-5f);
  int b = row / LSEQ;
  const float* sc = scsh + (size_t)b * 2 * HIDN;
  const float* sh = sc + HIDN;
  for (int c = tid; c < HIDN; c += 128)
    out[(size_t)row * HIDN + c] =
        (_Float16)((x[c] - mean) * rstd * (1.0f + sc[c]) + sh[c]);
}

// Final LayerNorm with affine (fn_g, fn_b) -> f32
__global__ __launch_bounds__(128) void k_lnfinal(const float* __restrict__ h,
                                                 const float* __restrict__ g,
                                                 const float* __restrict__ be,
                                                 float* __restrict__ out) {
  __shared__ float rs[128], rq[128];
  int row = blockIdx.x, tid = threadIdx.x;
  const float* x = h + (size_t)row * HIDN;
  float s = 0.0f, s2 = 0.0f;
  for (int c = tid; c < HIDN; c += 128) { float v = x[c]; s += v; s2 += v * v; }
  rs[tid] = s; rq[tid] = s2;
  __syncthreads();
  for (int o = 64; o > 0; o >>= 1) {
    if (tid < o) { rs[tid] += rs[tid + o]; rq[tid] += rq[tid + o]; }
    __syncthreads();
  }
  float mean = rs[0] / (float)HIDN;
  float var = rq[0] / (float)HIDN - mean * mean;
  float rstd = rsqrtf(var + 1e-5f);
  for (int c = tid; c < HIDN; c += 128)
    out[(size_t)row * HIDN + c] = (x[c] - mean) * rstd * g[c] + be[c];
}

// Output heads: node_out then edge_out, concatenated flat into d_out
__global__ void k_heads(const float* __restrict__ nf,
                        const float* __restrict__ on_w, const float* __restrict__ on_b,
                        const float* __restrict__ oe_w, const float* __restrict__ oe_b,
                        float* __restrict__ out) {
  const int NODE_TOT = BB * NATOM * ATYPES;
  const int EDGE_TOT = BB * NATOM * NATOM * ETYPES;
  int idx = blockIdx.x * blockDim.x + threadIdx.x;
  if (idx >= NODE_TOT + EDGE_TOT) return;
  if (idx < NODE_TOT) {
    int a = idx % ATYPES;
    int i = (idx / ATYPES) % NATOM;
    int b = idx / (ATYPES * NATOM);
    const float* np = nf + ((size_t)b * LSEQ + i) * HIDN;
    float acc = on_b[a];
    for (int k = 0; k < HIDN; ++k) acc += np[k] * on_w[k * ATYPES + a];
    out[idx] = acc;
  } else {
    int j = idx - NODE_TOT;
    int e = j % ETYPES;
    int p = (j / ETYPES) % (NATOM * NATOM);
    int b = j / (ETYPES * NATOM * NATOM);
    const float* np = nf + ((size_t)b * LSEQ + NATOM + p) * HIDN;
    float acc = oe_b[e];
    for (int k = 0; k < HIDN; ++k) acc += np[k] * oe_w[k * ETYPES + e];
    out[idx] = acc;
  }
}

// ---------------------------------------------------------------------------
extern "C" void kernel_launch(void* const* d_in, const int* in_sizes, int n_in,
                              void* d_out, int out_size, void* d_ws, size_t ws_size,
                              hipStream_t stream) {
  const float* x      = (const float*)d_in[0];
  const float* adj    = (const float*)d_in[1];
  const float* t      = (const float*)d_in[2];
  const float* node_w = (const float*)d_in[3];
  const float* node_b = (const float*)d_in[4];
  const float* edge_w = (const float*)d_in[5];
  const float* edge_b = (const float*)d_in[6];
  const float* pos    = (const float*)d_in[7];
  const float* t_w1   = (const float*)d_in[8];
  const float* t_b1   = (const float*)d_in[9];
  const float* t_w2   = (const float*)d_in[10];
  const float* t_b2   = (const float*)d_in[11];
  const float* ada1_w = (const float*)d_in[12];
  const float* ada1_b = (const float*)d_in[13];
  const float* ada2_w = (const float*)d_in[14];
  const float* ada2_b = (const float*)d_in[15];
  const float* q_w = (const float*)d_in[16];  const float* q_b = (const float*)d_in[17];
  const float* k_w = (const float*)d_in[18];  const float* k_b = (const float*)d_in[19];
  const float* v_w = (const float*)d_in[20];  const float* v_b = (const float*)d_in[21];
  const float* o_w = (const float*)d_in[22];  const float* o_b = (const float*)d_in[23];
  const float* m1_w = (const float*)d_in[24]; const float* m1_b = (const float*)d_in[25];
  const float* m2_w = (const float*)d_in[26]; const float* m2_b = (const float*)d_in[27];
  const float* fn_g = (const float*)d_in[28]; const float* fn_b = (const float*)d_in[29];
  const float* on_w = (const float*)d_in[30]; const float* on_b = (const float*)d_in[31];
  const float* oe_w = (const float*)d_in[32]; const float* oe_b = (const float*)d_in[33];
  float* out = (float*)d_out;

  // ---- workspace carve-up ----
  size_t off = 0;
  auto alloc = [&](size_t bytes) -> void* {
    void* p = (char*)d_ws + off;
    off += (bytes + 255) & ~(size_t)255;
    return p;
  };
  float*     h     = (float*)alloc((size_t)ROWS * HIDN * 4);
  _Float16*  n16   = (_Float16*)alloc((size_t)ROWS * HIDN * 2);
  _Float16*  q16   = (_Float16*)alloc((size_t)ROWS * HIDN * 2);
  _Float16*  k16   = (_Float16*)alloc((size_t)ROWS * HIDN * 2);
  _Float16*  v16   = (_Float16*)alloc((size_t)ROWS * HIDN * 2);
  _Float16*  a16   = (_Float16*)alloc((size_t)ROWS * HIDN * 2);
  _Float16*  mid16 = (_Float16*)alloc((size_t)ROWS * MLPN * 2);
  float*     nfin  = (float*)alloc((size_t)ROWS * HIDN * 4);
  float*     temb  = (float*)alloc((size_t)BB * HIDN * 4);
  float*     hmid  = (float*)alloc((size_t)BB * MLPN * 4);
  float*     cond  = (float*)alloc((size_t)BB * HIDN * 4);
  float*     scsh  = (float*)alloc((size_t)NLYR * 2 * BB * 2 * HIDN * 4);
  // packed weights (B-fragment layout)
  const size_t SQ = (size_t)HIDN * HIDN;          // 147456
  const size_t SM = (size_t)HIDN * MLPN;          // 589824
  const size_t PER_LAYER = 4 * SQ + 2 * SM;       // 1769472 halves
  _Float16* pw = (_Float16*)alloc(PER_LAYER * NLYR * 2);

  // ---- pack all GEMM weights into fragment layout ----
  for (int i = 0; i < NLYR; ++i) {
    _Float16* base = pw + (size_t)i * PER_LAYER;
    hipLaunchKernelGGL(k_pack, dim3(HIDN / 32, HIDN / 16), dim3(32), 0, stream,
                       q_w + (size_t)i * SQ, base, HIDN);
    hipLaunchKernelGGL(k_pack, dim3(HIDN / 32, HIDN / 16), dim3(32), 0, stream,
                       k_w + (size_t)i * SQ, base + SQ, HIDN);
    hipLaunchKernelGGL(k_pack, dim3(HIDN / 32, HIDN / 16), dim3(32), 0, stream,
                       v_w + (size_t)i * SQ, base + 2 * SQ, HIDN);
    hipLaunchKernelGGL(k_pack, dim3(HIDN / 32, HIDN / 16), dim3(32), 0, stream,
                       o_w + (size_t)i * SQ, base + 3 * SQ, HIDN);
    hipLaunchKernelGGL(k_pack, dim3(HIDN / 32, MLPN / 16), dim3(32), 0, stream,
                       m1_w + (size_t)i * SM, base + 4 * SQ, MLPN);
    hipLaunchKernelGGL(k_pack, dim3(MLPN / 32, HIDN / 16), dim3(32), 0, stream,
                       m2_w + (size_t)i * SM, base + 4 * SQ + SM, HIDN);
  }

  // ---- embeddings + timestep conditioning ----
  {
    int tot = ROWS * HIDN;
    hipLaunchKernelGGL(k_embed, dim3((tot + 255) / 256), dim3(256), 0, stream,
                       x, adj, node_w, node_b, edge_w, edge_b, pos, h);
  }
  hipLaunchKernelGGL(k_temb, dim3((BB * HIDN + 127) / 128), dim3(128), 0, stream, t, temb);
  hipLaunchKernelGGL(k_tmlp1, dim3((BB * MLPN + 127) / 128), dim3(128), 0, stream,
                     temb, t_w1, t_b1, hmid);
  hipLaunchKernelGGL(k_tmlp2, dim3((BB * HIDN + 127) / 128), dim3(128), 0, stream,
                     hmid, t_w2, t_b2, cond);
  {
    int tot = NLYR * 2 * BB * 2 * HIDN;
    hipLaunchKernelGGL(k_ada, dim3((tot + 127) / 128), dim3(128), 0, stream,
                       cond, ada1_w, ada1_b, ada2_w, ada2_b, scsh);
  }

  const dim3 gemm_sq(HIDN / 64, ROWS / 32);   // 6 x 147
  const dim3 gemm_m1(MLPN / 64, ROWS / 32);   // 24 x 147

  for (int i = 0; i < NLYR; ++i) {
    _Float16* base = pw + (size_t)i * PER_LAYER;
    const float* sc1 = scsh + (size_t)(i * 2 + 0) * BB * 2 * HIDN;
    const float* sc2 = scsh + (size_t)(i * 2 + 1) * BB * 2 * HIDN;

    // AdaLN-1 + attention
    hipLaunchKernelGGL(k_lnmod, dim3(ROWS), dim3(128), 0, stream, h, sc1, n16);
    hipLaunchKernelGGL(k_gemm, gemm_sq, dim3(32), 0, stream,
                       n16, base, q_b + (size_t)i * HIDN, (const float*)nullptr,
                       (float*)nullptr, q16, HIDN, HIDN, 0);
    hipLaunchKernelGGL(k_gemm, gemm_sq, dim3(32), 0, stream,
                       n16, base + SQ, k_b + (size_t)i * HIDN, (const float*)nullptr,
                       (float*)nullptr, k16, HIDN, HIDN, 0);
    hipLaunchKernelGGL(k_gemm, gemm_sq, dim3(32), 0, stream,
                       n16, base + 2 * SQ, v_b + (size_t)i * HIDN, (const float*)nullptr,
                       (float*)nullptr, v16, HIDN, HIDN, 0);
    hipLaunchKernelGGL(k_attn, dim3(LSEQ / 16, NHEAD, BB), dim3(32), 0, stream,
                       q16, k16, v16, a16);
    hipLaunchKernelGGL(k_gemm, gemm_sq, dim3(32), 0, stream,
                       a16, base + 3 * SQ, o_b + (size_t)i * HIDN, h, h,
                       (_Float16*)nullptr, HIDN, HIDN, 1);

    // AdaLN-2 + MLP
    hipLaunchKernelGGL(k_lnmod, dim3(ROWS), dim3(128), 0, stream, h, sc2, n16);
    hipLaunchKernelGGL(k_gemm, gemm_m1, dim3(32), 0, stream,
                       n16, base + 4 * SQ, m1_b + (size_t)i * MLPN, (const float*)nullptr,
                       (float*)nullptr, mid16, HIDN, MLPN, 2);
    hipLaunchKernelGGL(k_gemm, gemm_sq, dim3(32), 0, stream,
                       mid16, base + 4 * SQ + SM, m2_b + (size_t)i * HIDN, h, h,
                       (_Float16*)nullptr, MLPN, HIDN, 1);
  }

  // ---- final norm + heads ----
  hipLaunchKernelGGL(k_lnfinal, dim3(ROWS), dim3(128), 0, stream, h, fn_g, fn_b, nfin);
  {
    int tot = BB * NATOM * ATYPES + BB * NATOM * NATOM * ETYPES;
    hipLaunchKernelGGL(k_heads, dim3((tot + 255) / 256), dim3(256), 0, stream,
                       nfin, on_w, on_b, oe_w, oe_b, out);
  }
}
